// BoundingBoxRegressor_58317065945916
// MI455X (gfx1250) — matrix operand order repaired and checked by reading, not compiled, not run on hardware
//
#include <hip/hip_runtime.h>
#include <hip/hip_bf16.h>

#define DI __device__ __forceinline__

typedef __attribute__((ext_vector_type(16))) __bf16        v16bf;
typedef __attribute__((ext_vector_type(8)))  float         v8f;
typedef __attribute__((ext_vector_type(4)))  unsigned int  u32x4;

union FragB16 { v16bf v; u32x4 q[2]; };

// Dimensions
// B=32, L=196 (pad 224), D=1024, C=36 (pad 48), H=256, C*H=9216
#define B_  32
#define L_  196
#define LP  224
#define D_  1024
#define C_  36
#define CP  48
#define H_  256
#define CH  9216

DI unsigned short f32_to_bf16(float f) {
  unsigned int u = __float_as_uint(f);
  u += 0x7FFFu + ((u >> 16) & 1u);           // round-to-nearest-even
  return (unsigned short)(u >> 16);
}

// One wave computes one 16x16 f32 tile: acc += A(16xK) * Bt(16xK)^T.
// A row-major [16 x lda], Bt row-major [16 x ldb] (i.e. B stored N-major).
// K must be a multiple of 32; pointers 16B aligned; lda/ldb multiples of 8.
DI void wmma_loop(const unsigned short* __restrict__ A, int lda,
                  const unsigned short* __restrict__ Bt, int ldb,
                  int K, int lane, v8f& acc) {
  const int half = lane >> 4;   // lane group 0/1
  const int idx  = lane & 15;   // A row m / B column n
  const unsigned short* a0 = A  + idx * lda + half * 8;   // K = k + 8*half + 0..7 / +16
  const unsigned short* b0 = Bt + idx * ldb + half * 8;   // K = k + 8*half + 0..7 / +16
  for (int k = 0; k < K; k += 32) {
    FragB16 fa, fb;
    fa.q[0] = *(const u32x4*)(a0 + k);
    fa.q[1] = *(const u32x4*)(a0 + k + 16);
    fb.q[0] = *(const u32x4*)(b0 + k);
    fb.q[1] = *(const u32x4*)(b0 + k + 16);
    acc = __builtin_amdgcn_wmma_f32_16x16x32_bf16(false, fa.v, false, fb.v,
                                                  (short)0, acc, false, false);
  }
}

// ---------------- conversion / transpose ----------------

__global__ void k_convert(const float* __restrict__ src,
                          unsigned short* __restrict__ dst, int n) {
  int i = blockIdx.x * blockDim.x + threadIdx.x;
  if (i < n) dst[i] = f32_to_bf16(src[i]);
}

// local [32,196,1024] f32 -> [32,224,1024] bf16 (pad rows left as-is; outputs
// computed from them are discarded by store guards)
__global__ void k_convert_local(const float* __restrict__ src,
                                unsigned short* __restrict__ dst) {
  int i = blockIdx.x * blockDim.x + threadIdx.x;
  const int n = B_ * L_ * D_;
  if (i < n) {
    int b = i / (L_ * D_);
    int rem = i - b * (L_ * D_);
    dst[b * (LP * D_) + rem] = f32_to_bf16(src[i]);
  }
}

// dst[c, r] = (r < R) ? src[r, c] : 0   for r < Rpad, c < Cc.  32x32 LDS tiles.
__global__ void k_transpose_bf16(const float* __restrict__ src,
                                 unsigned short* __restrict__ dst,
                                 int R, int Cc, int Rpad,
                                 long srcStride, long dstStride) {
  __shared__ float tile[32][33];
  const float* s = src + srcStride * blockIdx.z;
  unsigned short* d = dst + dstStride * blockIdx.z;
  int c0 = blockIdx.x * 32, r0 = blockIdx.y * 32;
  int tx = threadIdx.x, ty = threadIdx.y;  // 32 x 8
#pragma unroll
  for (int i = 0; i < 4; ++i) {
    int r = r0 + ty + i * 8, c = c0 + tx;
    float v = 0.0f;
    if (r < R && c < Cc) v = s[(long)r * Cc + c];
    tile[ty + i * 8][tx] = v;
  }
  __syncthreads();
#pragma unroll
  for (int i = 0; i < 4; ++i) {
    int dr = c0 + ty + i * 8;  // dst row = src col
    int dc = r0 + tx;          // dst col = src row (zero padded to Rpad)
    if (dr < Cc && dc < Rpad)
      d[(long)dr * Rpad + dc] = f32_to_bf16(tile[tx][ty + i * 8]);
  }
}

// ---------------- GEMM stages ----------------

// xg[b, n] = gap[b,:] @ Wg[:, n] + bg[n]  -> bf16 [32, 9216]
__global__ void k_gemm_xg(const unsigned short* __restrict__ gapb,
                          const unsigned short* __restrict__ WgT,
                          const float* __restrict__ bg,
                          unsigned short* __restrict__ xg) {
  int gw = (blockIdx.x * blockDim.x + threadIdx.x) >> 5;
  int lane = threadIdx.x & 31;
  const int MT = 2, NT = CH / 16;
  if (gw >= MT * NT) return;
  int mt = gw % MT, nt = gw / MT;
  v8f acc = {};
  wmma_loop(gapb + mt * 16 * D_, D_, WgT + nt * 16 * D_, D_, D_, lane, acc);
  int n = nt * 16 + (lane & 15);
  int half = lane >> 4;
  float bias = bg[n];
#pragma unroll
  for (int r = 0; r < 8; ++r) {
    int m = mt * 16 + r + 8 * half;
    xg[m * CH + n] = f32_to_bf16(acc[r] + bias);
  }
}

// v[b, c, d] = sum_h xg[b, cH+h] * Wl[d, cH+h]   -> bf16 [32, 48, 1024]
__global__ void k_gemm_v(const unsigned short* __restrict__ xg,
                         const unsigned short* __restrict__ Wlb,
                         unsigned short* __restrict__ vb) {
  int gw = (blockIdx.x * blockDim.x + threadIdx.x) >> 5;
  int lane = threadIdx.x & 31;
  const int TPC = 2 * (D_ / 16);                 // 128 tiles per class
  if (gw >= C_ * TPC) return;
  int c = gw / TPC, t = gw % TPC;
  int mt = t & 1, nt = t >> 1;
  v8f acc = {};
  wmma_loop(xg + mt * 16 * CH + c * H_, CH,
            Wlb + nt * 16 * CH + c * H_, CH, H_, lane, acc);
  int n = nt * 16 + (lane & 15);
  int half = lane >> 4;
#pragma unroll
  for (int r = 0; r < 8; ++r) {
    int m = mt * 16 + r + 8 * half;               // b
    vb[(m * CP + c) * D_ + n] = f32_to_bf16(acc[r]);
  }
}

// dot[b, c, l] = local[b,l,:] . v[b,c,:]  -> f32 [32, 36, 224]
__global__ void k_gemm_dot(const unsigned short* __restrict__ localb,
                           const unsigned short* __restrict__ vb,
                           float* __restrict__ dotb) {
  int gw = (blockIdx.x * blockDim.x + threadIdx.x) >> 5;
  int lane = threadIdx.x & 31;
  const int MT = LP / 16, NT = CP / 16, TPB = MT * NT;  // 14 * 3
  if (gw >= B_ * TPB) return;
  int b = gw / TPB, t = gw % TPB;
  int mt = t % MT, nt = t / MT;
  v8f acc = {};
  wmma_loop(localb + b * (LP * D_) + mt * 16 * D_, D_,
            vb + b * (CP * D_) + nt * 16 * D_, D_, D_, lane, acc);
  int n = nt * 16 + (lane & 15);                  // c
  int half = lane >> 4;
  if (n < C_) {
#pragma unroll
    for (int r = 0; r < 8; ++r) {
      int m = mt * 16 + r + 8 * half;             // l
      if (m < L_) dotb[(b * C_ + n) * LP + m] = acc[r];
    }
  }
}

// softmax over l per (b,c) row; writes bf16 attn [32, 48, 224] zero-padded
__global__ void k_softmax(const float* __restrict__ dotb,
                          unsigned short* __restrict__ attn) {
  int gw = (blockIdx.x * blockDim.x + threadIdx.x) >> 5;
  int lane = threadIdx.x & 31;
  if (gw >= B_ * CP) return;
  int b = gw / CP, c = gw % CP;
  unsigned short* arow = attn + (b * CP + c) * LP;
  if (c >= C_) {                                   // padded class rows -> 0
    for (int l = lane; l < LP; l += 32) arow[l] = 0;
    return;
  }
  const float* drow = dotb + (b * C_ + c) * LP;
  float mx = -3.0e38f;
  for (int l = lane; l < L_; l += 32) mx = fmaxf(mx, drow[l]);
  for (int m = 16; m > 0; m >>= 1) mx = fmaxf(mx, __shfl_xor(mx, m, 32));
  float vals[7];
  float sum = 0.0f;
  int cnt = 0;
  for (int l = lane; l < L_; l += 32) { float e = __expf(drow[l] - mx); vals[cnt++] = e; sum += e; }
  for (int m = 16; m > 0; m >>= 1) sum += __shfl_xor(sum, m, 32);
  float inv = 1.0f / sum;
  cnt = 0;
  for (int l = lane; l < L_; l += 32) arow[l] = f32_to_bf16(vals[cnt++] * inv);
  for (int l = L_ + lane; l < LP; l += 32) arow[l] = 0;   // padded K -> 0
}

// p[b, c, d] = sum_l attn[b,c,l] * local[b,l,d]  -> bf16 [32, 36, 1024]
__global__ void k_gemm_p(const unsigned short* __restrict__ attn,
                         const unsigned short* __restrict__ localT,
                         unsigned short* __restrict__ Pb) {
  int gw = (blockIdx.x * blockDim.x + threadIdx.x) >> 5;
  int lane = threadIdx.x & 31;
  const int MT = CP / 16, NT = D_ / 16, TPB = MT * NT;   // 3 * 64
  if (gw >= B_ * TPB) return;
  int b = gw / TPB, t = gw % TPB;
  int mt = t % MT, nt = t / MT;
  v8f acc = {};
  wmma_loop(attn + b * (CP * LP) + mt * 16 * LP, LP,
            localT + b * (D_ * LP) + nt * 16 * LP, LP, LP, lane, acc);
  int n = nt * 16 + (lane & 15);                  // d
  int half = lane >> 4;
#pragma unroll
  for (int r = 0; r < 8; ++r) {
    int m = mt * 16 + r + 8 * half;               // c
    if (m < C_) Pb[(b * C_ + m) * D_ + n] = f32_to_bf16(acc[r]);
  }
}

// ws[b, c, h] = p[b,c,:] @ Wl[:, cH+h] + bl[cH+h]  -> bf16 [1152, 256]
__global__ void k_gemm_ws(const unsigned short* __restrict__ Pb,
                          const unsigned short* __restrict__ WlT,
                          const float* __restrict__ bl,
                          unsigned short* __restrict__ wsb) {
  int gw = (blockIdx.x * blockDim.x + threadIdx.x) >> 5;
  int lane = threadIdx.x & 31;
  const int TPC = 2 * (H_ / 16);                  // 32 tiles per class
  if (gw >= C_ * TPC) return;
  int c = gw / TPC, t = gw % TPC;
  int mt = t & 1, nt = t >> 1;
  v8f acc = {};
  wmma_loop(Pb + mt * 16 * (C_ * D_) + c * D_, C_ * D_,
            WlT + (c * H_ + nt * 16) * D_, D_, D_, lane, acc);
  int n = nt * 16 + (lane & 15);                  // h
  int half = lane >> 4;
  float bias = bl[c * H_ + n];
#pragma unroll
  for (int r = 0; r < 8; ++r) {
    int m = mt * 16 + r + 8 * half;               // b
    wsb[(m * C_ + c) * H_ + n] = f32_to_bf16(acc[r] + bias);
  }
}

// hid = relu(ws @ Wh + bh)  -> f32 [1152, 256]
__global__ void k_gemm_hid(const unsigned short* __restrict__ wsb,
                           const unsigned short* __restrict__ WhT,
                           const float* __restrict__ bh,
                           float* __restrict__ hid) {
  int gw = (blockIdx.x * blockDim.x + threadIdx.x) >> 5;
  int lane = threadIdx.x & 31;
  const int MT = (B_ * C_) / 16, NT = H_ / 16;    // 72 * 16
  if (gw >= MT * NT) return;
  int mt = gw % MT, nt = gw / MT;
  v8f acc = {};
  wmma_loop(wsb + mt * 16 * H_, H_, WhT + nt * 16 * H_, H_, H_, lane, acc);
  int n = nt * 16 + (lane & 15);
  int half = lane >> 4;
  float bias = bh[n];
#pragma unroll
  for (int r = 0; r < 8; ++r) {
    int m = mt * 16 + r + 8 * half;
    float x = acc[r] + bias;
    hid[m * H_ + n] = x > 0.0f ? x : 0.0f;
  }
}

// heads: bbox [32,144] then presence [32,36], concatenated in d_out
__global__ void k_heads(const float* __restrict__ hid,
                        const float* __restrict__ Wr, const float* __restrict__ br,
                        const float* __restrict__ Wc, const float* __restrict__ bc,
                        float* __restrict__ out) {
  int gw = (blockIdx.x * blockDim.x + threadIdx.x) >> 5;
  int lane = threadIdx.x & 31;
  if (gw >= B_ * C_) return;
  const float* row = hid + gw * H_;
  float a0 = 0, a1 = 0, a2 = 0, a3 = 0, a4 = 0;
  for (int h = lane; h < H_; h += 32) {
    float v = row[h];
    a0 += v * Wr[h * 4 + 0];
    a1 += v * Wr[h * 4 + 1];
    a2 += v * Wr[h * 4 + 2];
    a3 += v * Wr[h * 4 + 3];
    a4 += v * Wc[h];
  }
  for (int m = 16; m > 0; m >>= 1) {
    a0 += __shfl_xor(a0, m, 32);
    a1 += __shfl_xor(a1, m, 32);
    a2 += __shfl_xor(a2, m, 32);
    a3 += __shfl_xor(a3, m, 32);
    a4 += __shfl_xor(a4, m, 32);
  }
  if (lane == 0) {
    int b = gw / C_, c = gw % C_;
    out[b * (C_ * 4) + c * 4 + 0] = a0 + br[0];
    out[b * (C_ * 4) + c * 4 + 1] = a1 + br[1];
    out[b * (C_ * 4) + c * 4 + 2] = a2 + br[2];
    out[b * (C_ * 4) + c * 4 + 3] = a3 + br[3];
    out[B_ * C_ * 4 + b * C_ + c] = a4 + bc[0];
  }
}

extern "C" void kernel_launch(void* const* d_in, const int* in_sizes, int n_in,
                              void* d_out, int out_size, void* d_ws, size_t ws_size,
                              hipStream_t stream) {
  (void)in_sizes; (void)n_in; (void)out_size; (void)ws_size;
  const float* local = (const float*)d_in[0];
  const float* gap   = (const float*)d_in[1];
  const float* Wl    = (const float*)d_in[2];
  const float* bl    = (const float*)d_in[3];
  const float* Wg    = (const float*)d_in[4];
  const float* bg    = (const float*)d_in[5];
  const float* Wh    = (const float*)d_in[6];
  const float* bh    = (const float*)d_in[7];
  const float* Wr    = (const float*)d_in[8];
  const float* br    = (const float*)d_in[9];
  const float* Wc    = (const float*)d_in[10];
  const float* bc    = (const float*)d_in[11];
  float* out = (float*)d_out;

  char* ws = (char*)d_ws;
  size_t off = 0;
  auto alloc = [&](size_t bytes) -> char* {
    char* p = ws + off;
    off = (off + bytes + 255) & ~(size_t)255;
    return p;
  };
  unsigned short* localb = (unsigned short*)alloc((size_t)B_ * LP * D_ * 2);
  unsigned short* localT = (unsigned short*)alloc((size_t)B_ * D_ * LP * 2);
  unsigned short* gapb   = (unsigned short*)alloc((size_t)B_ * D_ * 2);
  unsigned short* Wlb    = (unsigned short*)alloc((size_t)D_ * CH * 2);
  unsigned short* WlT    = (unsigned short*)alloc((size_t)CH * D_ * 2);
  unsigned short* WgT    = (unsigned short*)alloc((size_t)CH * D_ * 2);
  unsigned short* WhT    = (unsigned short*)alloc((size_t)H_ * H_ * 2);
  unsigned short* xgb    = (unsigned short*)alloc((size_t)B_ * CH * 2);
  unsigned short* vb     = (unsigned short*)alloc((size_t)B_ * CP * D_ * 2);
  float*          dotb   = (float*)alloc((size_t)B_ * C_ * LP * 4);
  unsigned short* attn   = (unsigned short*)alloc((size_t)B_ * CP * LP * 2);
  unsigned short* Pb     = (unsigned short*)alloc((size_t)B_ * C_ * D_ * 2);
  unsigned short* wsb    = (unsigned short*)alloc((size_t)B_ * C_ * H_ * 2);
  float*          hid    = (float*)alloc((size_t)B_ * C_ * H_ * 4);

  const int TB = 256;
  auto blocksFor = [](int n, int per) { return (n + per - 1) / per; };

  // 1) conversions
  k_convert<<<blocksFor(B_ * D_, TB), TB, 0, stream>>>(gap, gapb, B_ * D_);
  k_convert_local<<<blocksFor(B_ * L_ * D_, TB), TB, 0, stream>>>(local, localb);
  k_convert<<<blocksFor(D_ * CH, TB), TB, 0, stream>>>(Wl, Wlb, D_ * CH);
  // 2) transposes (LDS tiled)
  dim3 tb(32, 8);
  k_transpose_bf16<<<dim3(CH / 32, D_ / 32, 1), tb, 0, stream>>>(Wl, WlT, D_, CH, D_, 0, 0);
  k_transpose_bf16<<<dim3(CH / 32, D_ / 32, 1), tb, 0, stream>>>(Wg, WgT, D_, CH, D_, 0, 0);
  k_transpose_bf16<<<dim3(H_ / 32, H_ / 32, 1), tb, 0, stream>>>(Wh, WhT, H_, H_, H_, 0, 0);
  k_transpose_bf16<<<dim3(D_ / 32, LP / 32, B_), tb, 0, stream>>>(
      local, localT, L_, D_, LP, (long)L_ * D_, (long)D_ * LP);

  // 3) xg = gap@Wg + bg        (2 * 576 wave-tiles)
  k_gemm_xg<<<blocksFor(2 * (CH / 16), 8), TB, 0, stream>>>(gapb, WgT, bg, xgb);
  // 4) v = per-class Wl_c @ xg  (36 * 128 wave-tiles)
  k_gemm_v<<<blocksFor(C_ * 2 * (D_ / 16), 8), TB, 0, stream>>>(xgb, Wlb, vb);
  // 5) dot = local @ v^T        (32 * 42 wave-tiles)
  k_gemm_dot<<<blocksFor(B_ * (LP / 16) * (CP / 16), 8), TB, 0, stream>>>(localb, vb, dotb);
  // 6) softmax                  (32*48 wave-rows)
  k_softmax<<<blocksFor(B_ * CP, 8), TB, 0, stream>>>(dotb, attn);
  // 7) p = attn @ local         (32 * 192 wave-tiles)
  k_gemm_p<<<blocksFor(B_ * (CP / 16) * (D_ / 16), 8), TB, 0, stream>>>(attn, localT, Pb);
  // 8) ws = p @ Wl_c + bl       (36 * 32 wave-tiles)
  k_gemm_ws<<<blocksFor(C_ * 2 * (H_ / 16), 8), TB, 0, stream>>>(Pb, WlT, bl, wsb);
  // 9) hid = relu(ws@Wh + bh)   (72 * 16 wave-tiles)
  k_gemm_hid<<<blocksFor((B_ * C_ / 16) * (H_ / 16), 8), TB, 0, stream>>>(wsb, WhT, bh, hid);
  // 10) heads
  k_heads<<<blocksFor(B_ * C_, 8), TB, 0, stream>>>(hid, Wr, br, Wc, bc, out);
}
